// LSTMSimulator_42485816492218
// MI455X (gfx1250) — compile-verified, matrix operand order checked by reading, not run on hardware
//
#include <hip/hip_runtime.h>

#define H    51
#define H4   204
#define NT   13        // 13 N-tiles of 16 -> 208 padded gate columns
#define NP   208
#define ROWS 64        // batch rows per workgroup
#define TT   2048
#define BB   8192
#define THREADS 256    // 8 wave32 waves: 2 waves share each 16-row strip, split N-tiles

typedef _Float16 v8h  __attribute__((ext_vector_type(8)));
typedef _Float16 v16h __attribute__((ext_vector_type(16)));
typedef float    v8f  __attribute__((ext_vector_type(8)));

// CDNA5 hardware transcendental v_tanh_f32 when available.
#if __has_builtin(__builtin_amdgcn_tanhf)
__device__ __forceinline__ float ftanh(float x) { return __builtin_amdgcn_tanhf(x); }
#elif __has_builtin(__builtin_amdgcn_tanh_f32)
__device__ __forceinline__ float ftanh(float x) { return __builtin_amdgcn_tanh_f32(x); }
#else
__device__ __forceinline__ float ftanh(float x) { return tanhf(x); }
#endif
// sigmoid(x) = 0.5 + 0.5*tanh(x/2): one transcendental + FMAs
__device__ __forceinline__ float sigf(float x) { return 0.5f + 0.5f * ftanh(0.5f * x); }

// CNT output tiles, stride-2 interleaved starting at nt0.
template<int CNT, int KB>
__device__ __forceinline__ void mm_tiles(int nt0, int lane, int ccol, int mbase,
                                         const _Float16 Wp[][KB][32][16],
                                         const v16h* a, _Float16 (*gates)[72])
{
#pragma unroll
  for (int i = 0; i < CNT; ++i) {
    const int nt = nt0 + 2 * i;
    v8f acc = {};
#pragma unroll
    for (int kb = 0; kb < KB; ++kb)
      acc = __builtin_amdgcn_wmma_f32_16x16x32_f16(false, a[kb], false,
               *(const v16h*)&Wp[nt][kb][lane][0], (short)0, acc, false, false);
    const int col = nt * 16 + ccol;
#pragma unroll
    for (int rr = 0; rr < 4; ++rr) {   // row pairs -> b32; compiler fuses to cvt_pk + b128
      union { _Float16 h[2]; unsigned u; } p;
      p.h[0] = (_Float16)acc[2 * rr];
      p.h[1] = (_Float16)acc[2 * rr + 1];
      *(unsigned*)&gates[col][mbase + 2 * rr] = p.u;
    }
  }
}

__global__ __launch_bounds__(THREADS) void lstm2_wmma_kernel(
    const float* __restrict__ input,
    const float* __restrict__ W_ih1, const float* __restrict__ W_hh1,
    const float* __restrict__ b_ih1, const float* __restrict__ b_hh1,
    const float* __restrict__ W_ih2, const float* __restrict__ W_hh2,
    const float* __restrict__ b_ih2, const float* __restrict__ b_hh2,
    const float* __restrict__ W_lin, const float* __restrict__ b_lin,
    float* __restrict__ out)
{
  // LDS ~153 KB total -> two WGs can co-reside in a 320 KB WGP
  __shared__ _Float16 Abuf[ROWS][136];   // [row][k]: k 0..63 = h1(pad), 64..127 = h2(pad)
  __shared__ _Float16 gates[NP][72];     // column-major gate staging [col][row]
  __shared__ float    c1[ROWS][52];
  __shared__ float    c2[ROWS][52];
  __shared__ _Float16 W1p[NT][2][32][16]; // B-operand-packed W_hh1  (K=64)
  __shared__ _Float16 W2p[NT][4][32][16]; // B-operand-packed [W_ih2;W_hh2] (K=128)
  __shared__ float    wih1f[NP], b1s[NP], b2s[NP];
  __shared__ float    wlinf[64];
  __shared__ float    xbuf[ROWS];
  __shared__ float    outacc[ROWS];
  __shared__ float    blin;

  const int tid  = threadIdx.x;
  const int lane = tid & 31;
  const int wave = tid >> 5;
  const int row0 = blockIdx.x * ROWS;

  // ---------------- one-time setup: pack weights f32 -> f16 WMMA-B layout ----------------
  for (int idx = tid; idx < NT*2*32*16; idx += THREADS) {
    int e = idx & 15, ln = (idx >> 4) & 31, kb = (idx >> 9) & 1, nt = idx >> 10;
    int n = nt*16 + (ln & 15);
    int k = kb*32 + ((ln >> 4) << 3) + (e < 8 ? e : e + 8);
    float w = (n < H4 && k < H) ? W_hh1[n*H + k] : 0.0f;
    W1p[nt][kb][ln][e] = (_Float16)w;
  }
  for (int idx = tid; idx < NT*4*32*16; idx += THREADS) {
    int e = idx & 15, ln = (idx >> 4) & 31, kb = (idx >> 9) & 3, nt = idx >> 11;
    int n = nt*16 + (ln & 15);
    int k = kb*32 + ((ln >> 4) << 3) + (e < 8 ? e : e + 8);
    float w = 0.0f;
    if (n < H4) {
      if (k < 64) { if (k < H) w = W_ih2[n*H + k]; }
      else        { int kk = k - 64; if (kk < H) w = W_hh2[n*H + kk]; }
    }
    W2p[nt][kb][ln][e] = (_Float16)w;
  }
  for (int n = tid; n < NP; n += THREADS) {
    bool v = n < H4;
    wih1f[n] = v ? W_ih1[n] : 0.0f;
    b1s[n]   = v ? (b_ih1[n] + b_hh1[n]) : 0.0f;
    b2s[n]   = v ? (b_ih2[n] + b_hh2[n]) : 0.0f;
  }
  if (tid < 64) wlinf[tid] = (tid < H) ? W_lin[tid] : 0.0f;
  if (tid == 0) blin = b_lin[0];
  for (int idx = tid; idx < ROWS*52; idx += THREADS) {
    (&c1[0][0])[idx] = 0.0f;
    (&c2[0][0])[idx] = 0.0f;
  }
  for (int idx = tid; idx < ROWS*136; idx += THREADS)
    (&Abuf[0][0])[idx] = (_Float16)0.0f;
  __syncthreads();

  const int m0    = (wave & 3) << 4;         // wave's 16-row M strip (2 waves per strip)
  // N-tile interleave select, forced wave-uniform -> plain scalar branch, no saveexec
  const int half  = __builtin_amdgcn_readfirstlane(wave >> 2);
  const int arow  = m0 + (lane & 15);
  const int khalf = (lane >> 4) << 3;        // 0 or 8 (A/B K-half select)
  const int ccol  = lane & 15;
  const int mbase = m0 + ((lane >> 4) << 3); // C/D row base for this lane half
  const int erow  = tid & 63;                // elementwise row (4 threads per row)

  // software-pipelined x_t: fetch t+1 while step t computes (hides L2 latency)
  float xcur = 0.0f;
  if (tid < ROWS) xcur = input[(size_t)(row0 + tid) * TT];

  for (int t = 0; t < TT; ++t) {
    if (tid < ROWS) xbuf[tid] = xcur;
    __syncthreads();

    float xnxt = 0.0f;
    if (tid < ROWS && t + 1 < TT) {
      xnxt = input[(size_t)(row0 + tid) * TT + t + 1];
      if (t + 17 < TT)
        __builtin_prefetch(&input[(size_t)(row0 + tid) * TT + t + 17], 0, 0);
    }

    // ---- gates1 = h1 @ W_hh1^T  via v_wmma_f32_16x16x32_f16 (K=64) ----
    {
      v16h a[2];
#pragma unroll
      for (int kb = 0; kb < 2; ++kb) {
        v8h lo = *(const v8h*)&Abuf[arow][kb*32 + khalf];
        v8h hi = *(const v8h*)&Abuf[arow][kb*32 + khalf + 16];
        a[kb] = __builtin_shufflevector(lo, hi, 0,1,2,3,4,5,6,7,8,9,10,11,12,13,14,15);
      }
      if (half == 0) mm_tiles<7, 2>(0, lane, ccol, mbase, W1p, a, gates);
      else           mm_tiles<6, 2>(1, lane, ccol, mbase, W1p, a, gates);
    }
    __syncthreads();

    // ---- layer-1 elementwise: x*W_ih1 + biases, LSTM cell, h1_new -> Abuf f16 ----
    {
      float xv = xbuf[erow];
      for (int idx = tid; idx < ROWS*H; idx += THREADS) {
        int j = idx >> 6;
        float gi = (float)gates[j    ][erow] + xv*wih1f[j    ] + b1s[j    ];
        float gf = (float)gates[j+H  ][erow] + xv*wih1f[j+H  ] + b1s[j+H  ];
        float gg = (float)gates[j+2*H][erow] + xv*wih1f[j+2*H] + b1s[j+2*H];
        float go = (float)gates[j+3*H][erow] + xv*wih1f[j+3*H] + b1s[j+3*H];
        float cn = sigf(gf)*c1[erow][j] + sigf(gi)*ftanh(gg);
        c1[erow][j] = cn;
        Abuf[erow][j] = (_Float16)(sigf(go)*ftanh(cn));
      }
      if (tid < ROWS) outacc[tid] = 0.0f;
    }
    __syncthreads();

    // ---- gates2 = [h1_new | h2] @ [W_ih2 ; W_hh2]^T  (K=128) ----
    {
      v16h a[4];
#pragma unroll
      for (int kb = 0; kb < 4; ++kb) {
        v8h lo = *(const v8h*)&Abuf[arow][kb*32 + khalf];
        v8h hi = *(const v8h*)&Abuf[arow][kb*32 + khalf + 16];
        a[kb] = __builtin_shufflevector(lo, hi, 0,1,2,3,4,5,6,7,8,9,10,11,12,13,14,15);
      }
      if (half == 0) mm_tiles<7, 4>(0, lane, ccol, mbase, W2p, a, gates);
      else           mm_tiles<6, 4>(1, lane, ccol, mbase, W2p, a, gates);
    }
    __syncthreads();

    // ---- layer-2 elementwise + fused output projection partials ----
    {
      float osum = 0.0f;
      for (int idx = tid; idx < ROWS*H; idx += THREADS) {
        int j = idx >> 6;
        float gi = (float)gates[j    ][erow] + b2s[j    ];
        float gf = (float)gates[j+H  ][erow] + b2s[j+H  ];
        float gg = (float)gates[j+2*H][erow] + b2s[j+2*H];
        float go = (float)gates[j+3*H][erow] + b2s[j+3*H];
        float cn = sigf(gf)*c2[erow][j] + sigf(gi)*ftanh(gg);
        c2[erow][j] = cn;
        float hn = sigf(go)*ftanh(cn);
        Abuf[erow][64 + j] = (_Float16)hn;
        osum += hn * wlinf[j];
      }
      atomicAdd(&outacc[erow], osum);   // ds_add_f32, 4 partials per row
    }
    __syncthreads();

    if (tid < ROWS)
      out[(size_t)(row0 + tid) * TT + t] = outacc[tid] + blin;
    xcur = xnxt;
    // outacc re-zeroed only after two more barriers next iteration -> no extra sync needed
  }
}

extern "C" void kernel_launch(void* const* d_in, const int* in_sizes, int n_in,
                              void* d_out, int out_size, void* d_ws, size_t ws_size,
                              hipStream_t stream) {
  (void)in_sizes; (void)n_in; (void)out_size; (void)d_ws; (void)ws_size;
  const float* input = (const float*)d_in[0];
  const float* W_ih1 = (const float*)d_in[1];
  const float* W_hh1 = (const float*)d_in[2];
  const float* b_ih1 = (const float*)d_in[3];
  const float* b_hh1 = (const float*)d_in[4];
  const float* W_ih2 = (const float*)d_in[5];
  const float* W_hh2 = (const float*)d_in[6];
  const float* b_ih2 = (const float*)d_in[7];
  const float* b_hh2 = (const float*)d_in[8];
  const float* W_lin = (const float*)d_in[9];
  const float* b_lin = (const float*)d_in[10];
  float* out = (float*)d_out;

  dim3 grid(BB / ROWS);   // 128 persistent workgroups
  dim3 block(THREADS);    // 8 wave32 waves
  lstm2_wmma_kernel<<<grid, block, 0, stream>>>(
      input, W_ih1, W_hh1, b_ih1, b_hh1,
      W_ih2, W_hh2, b_ih2, b_hh2, W_lin, b_lin, out);
}
